// GroupAttention_2860448219592
// MI455X (gfx1250) — matrix operand order repaired and checked by reading, not compiled, not run on hardware
//
#include <hip/hip_runtime.h>
#include <hip/hip_bf16.h>

// ---------------- problem constants (match reference) ----------------
#define D_MODEL 512
#define BATCH   8
#define SEQ     1024
#define BS      (BATCH * SEQ)        // 8192 rows total
#define EPSV    1e-9f
#define INV_S   (1.0f / (float)SEQ)

#define USE_ASYNC_LDS 1              // gfx1250 async-to-LDS staging

// ---------------- WMMA types ----------------
typedef __attribute__((ext_vector_type(16))) __bf16 v16bf;
typedef __attribute__((ext_vector_type(8)))  float  v8f;

union ABf { unsigned int u[8]; v16bf v; };

__device__ __forceinline__ unsigned int f2bf(float f) {
    unsigned int x = __float_as_uint(f);
    unsigned int r = x + 0x7FFFu + ((x >> 16) & 1u);   // round-to-nearest-even
    return r >> 16;
}
__device__ __forceinline__ unsigned int pack2(float lo, float hi) {
    return f2bf(lo) | (f2bf(hi) << 16);
}

// =====================================================================
// Kernel 0: bulk f32 -> packed bf16 conversion (8 floats / thread).
// =====================================================================
__global__ __launch_bounds__(256) void cvt_f32_bf16(
    const float* __restrict__ src, uint4* __restrict__ dst)
{
    const size_t t = (size_t)blockIdx.x * 256 + threadIdx.x;
    const float4 a = ((const float4*)src)[2 * t + 0];
    const float4 b = ((const float4*)src)[2 * t + 1];
    uint4 o;
    o.x = pack2(a.x, a.y); o.y = pack2(a.z, a.w);
    o.z = pack2(b.x, b.y); o.w = pack2(b.z, b.w);
    dst[t] = o;
}

// =====================================================================
// Kernel 1: Q = ctx @ Wq^T + bq ; K = ctx @ Wk^T + bk   via bf16 WMMA.
// One block (8 waves) per 16-row M-tile; context tile read ONCE.
//  - double-buffered 1KB A tiles staged via global_load_async_to_lds_b128
//    (stage k+1 while computing k: async latency fully hidden)
//  - all 16 B-fragment b128 loads issued before the 8-WMMA burst
// =====================================================================
__global__ __launch_bounds__(256) void proj_qk_wmma(
    const unsigned short* __restrict__ ctxbf,
    const unsigned short* __restrict__ wqbf,
    const unsigned short* __restrict__ wkbf,
    const float* __restrict__ bq, const float* __restrict__ bk,
    float* __restrict__ Qo, float* __restrict__ Ko)
{
    __shared__ __align__(16) unsigned short sA[2][16 * 32];   // 2 x 1KB A tiles

    const int tid   = threadIdx.x;
    const int lane  = tid & 31;
    const int wave  = tid >> 5;
    const int mt    = blockIdx.x;          // M tile (0..511)
    const int m     = lane & 15;
    const int khalf = lane >> 4;

    // stage one 16x32 bf16 tile (64 x 16B chunks) into sA[buf]
    auto stage = [&](int buf, int k0) {
        if (tid < 64) {
            const unsigned short* gsrc =
                ctxbf + (size_t)(mt * 16 + (tid >> 2)) * D_MODEL + k0 + (tid & 3) * 8;
#if USE_ASYNC_LDS
            const unsigned ldsOff =
                (unsigned)(uintptr_t)(&sA[buf][0]) + (unsigned)(tid * 16);
            asm volatile("global_load_async_to_lds_b128 %0, %1, off"
                         :: "v"(ldsOff),
                            "v"((unsigned long long)(uintptr_t)gsrc)
                         : "memory");
#else
            const uint4 tmp = *(const uint4*)gsrc;
            *(uint4*)(&sA[buf][tid * 8]) = tmp;
#endif
        }
    };

    v8f accQ[4], accK[4];
    #pragma unroll
    for (int j = 0; j < 4; ++j) { accQ[j] = (v8f){}; accK[j] = (v8f){}; }

    stage(0, 0);
    int cur = 0;

    for (int k0 = 0; k0 < D_MODEL; k0 += 32) {
#if USE_ASYNC_LDS
        asm volatile("s_wait_asynccnt 0x0" ::: "memory");
#endif
        __syncthreads();                       // sA[cur] ready; sA[cur^1] free

        if (k0 + 32 < D_MODEL) stage(cur ^ 1, k0 + 32);

        // ---- A fragment: two ds_load_b128 per lane ----
        ABf a;
        {
            const uint4 a0 = *(const uint4*)(&sA[cur][m * 32 + khalf * 16]);
            const uint4 a1 = *(const uint4*)(&sA[cur][m * 32 + khalf * 16 + 8]);
            a.u[0] = a0.x; a.u[1] = a0.y; a.u[2] = a0.z; a.u[3] = a0.w;
            a.u[4] = a1.x; a.u[5] = a1.y; a.u[6] = a1.z; a.u[7] = a1.w;
        }

        // ---- preload ALL B fragments (16 x global_load_b128) ----
        ABf bqf[4], bkf[4];
        #pragma unroll
        for (int j = 0; j < 4; ++j) {
            const int ncol = (wave * 4 + j) * 16 + m;
            const size_t bbase = (size_t)ncol * D_MODEL + k0 + khalf * 16;
            const uint4 q0 = *(const uint4*)(wqbf + bbase);
            const uint4 q1 = *(const uint4*)(wqbf + bbase + 8);
            bqf[j].u[0] = q0.x; bqf[j].u[1] = q0.y; bqf[j].u[2] = q0.z; bqf[j].u[3] = q0.w;
            bqf[j].u[4] = q1.x; bqf[j].u[5] = q1.y; bqf[j].u[6] = q1.z; bqf[j].u[7] = q1.w;
            const uint4 c0 = *(const uint4*)(wkbf + bbase);
            const uint4 c1 = *(const uint4*)(wkbf + bbase + 8);
            bkf[j].u[0] = c0.x; bkf[j].u[1] = c0.y; bkf[j].u[2] = c0.z; bkf[j].u[3] = c0.w;
            bkf[j].u[4] = c1.x; bkf[j].u[5] = c1.y; bkf[j].u[6] = c1.z; bkf[j].u[7] = c1.w;
        }

        // ---- 8-WMMA burst (independent accumulators, shared A) ----
        #pragma unroll
        for (int j = 0; j < 4; ++j) {
            accQ[j] = __builtin_amdgcn_wmma_f32_16x16x32_bf16(
                          false, a.v, false, bqf[j].v, (short)0, accQ[j], false, false);
            accK[j] = __builtin_amdgcn_wmma_f32_16x16x32_bf16(
                          false, a.v, false, bkf[j].v, (short)0, accK[j], false, false);
        }

        cur ^= 1;
    }

    // ---- epilogue: add bias, store f32 Q/K ----
    #pragma unroll
    for (int j = 0; j < 4; ++j) {
        const int ncol  = (wave * 4 + j) * 16 + m;
        const float bqv = bq[ncol];
        const float bkv = bk[ncol];
        #pragma unroll
        for (int r = 0; r < 8; ++r) {
            const int row = mt * 16 + r + 8 * khalf;   // C/D layout
            const size_t off = (size_t)row * D_MODEL + ncol;
            Qo[off] = accQ[j][r] + bqv;
            Ko[off] = accK[j][r] + bkv;
        }
    }
}

// =====================================================================
// Kernel 2: per-row neighbor scores + 2-entry masked softmax.
// =====================================================================
__global__ __launch_bounds__(256) void row_stats(
    const float* __restrict__ Q, const float* __restrict__ K,
    const int*   __restrict__ eos,
    float* __restrict__ a_up, float* __restrict__ a_dn,
    float* __restrict__ unif)
{
    const int wave = threadIdx.x >> 5;
    const int lane = threadIdx.x & 31;
    const int rid  = blockIdx.x * 8 + wave;
    const int q    = rid & (SEQ - 1);

    const float* qrow = Q + (size_t)rid * D_MODEL;
    const float* kup  = K + (size_t)((q < SEQ - 1) ? rid + 1 : rid) * D_MODEL;
    const float* kdn  = K + (size_t)((q > 0)       ? rid - 1 : rid) * D_MODEL;

    float du = 0.f, dv = 0.f;
    #pragma unroll 4
    for (int d = lane; d < D_MODEL; d += 32) {
        const float qd = qrow[d];
        du += qd * kup[d];
        dv += qd * kdn[d];
    }
    #pragma unroll
    for (int off = 16; off; off >>= 1) {
        du += __shfl_xor(du, off, 32);
        dv += __shfl_xor(dv, off, 32);
    }

    if (lane == 0) {
        const float u = du * (1.0f / D_MODEL);
        const float v = dv * (1.0f / D_MODEL);
        const int b = rid >> 10;
        const size_t rowbase = (size_t)b * SEQ * SEQ + (size_t)q * SEQ;
        const int mu = (q < SEQ - 1) && (eos[rowbase + q + 1] != 0);
        const int md = (q > 0)       && (eos[rowbase + q - 1] != 0);
        float up = 0.f, dn = 0.f, un = 0.f;
        if (mu && md) {
            const float mx = fmaxf(u, v);
            const float eu = __expf(u - mx), ev = __expf(v - mx);
            const float inv = 1.0f / (eu + ev);
            up = eu * inv; dn = ev * inv;
        } else if (mu) up = 1.0f;
        else if (md)   dn = 1.0f;
        else           un = 1.0f;   // fully masked row -> uniform 1/S softmax
        a_up[rid] = up; a_dn[rid] = dn; unif[rid] = un;
    }
}

// =====================================================================
// Kernel 3: per-batch L[j] = log(neibor[j,j+1]+eps), exclusive scan P.
// =====================================================================
__global__ __launch_bounds__(1024) void scan_prefix(
    const float* __restrict__ a_up, const float* __restrict__ a_dn,
    const float* __restrict__ unif, const float* __restrict__ prior,
    float* __restrict__ P)
{
    __shared__ float sm[SEQ];
    const int b = blockIdx.x, j = threadIdx.x;
    const int rid = b * SEQ + j;

    float L = 0.f;
    if (j < SEQ - 1) {
        const float aup = (unif[rid]     > 0.5f) ? INV_S : a_up[rid];
        const float adn = (unif[rid + 1] > 0.5f) ? INV_S : a_dn[rid + 1];
        const float nb  = sqrtf(aup * adn + EPSV);
        const float pr  = prior[(size_t)b * SEQ * SEQ + (size_t)j * SEQ + (j + 1)];
        L = __logf(pr + (1.0f - pr) * nb + EPSV);
    }
    sm[j] = L;
    __syncthreads();
    for (int off = 1; off < SEQ; off <<= 1) {
        const float t = (j >= off) ? sm[j - off] : 0.f;
        __syncthreads();
        sm[j] += t;
        __syncthreads();
    }
    P[rid] = (j == 0) ? 0.f : sm[j - 1];
}

// =====================================================================
// Kernel 4: streaming write of both [B,S,S] outputs, float4-vectorized.
// =====================================================================
__global__ __launch_bounds__(256) void write_outputs(
    const float* __restrict__ prior,
    const float* __restrict__ a_up, const float* __restrict__ a_dn,
    const float* __restrict__ unif, const float* __restrict__ P,
    float* __restrict__ out)
{
    const size_t SS  = (size_t)SEQ * SEQ;
    const size_t vid = (size_t)blockIdx.x * blockDim.x + threadIdx.x;
    const size_t idx = vid * 4;
    const int b   = (int)(idx / SS);
    const size_t rem = idx - (size_t)b * SS;
    const int i  = (int)(rem / SEQ);
    const int k0 = (int)(rem & (SEQ - 1));
    const int rbase = b * SEQ;

    const float unif_i = unif[rbase + i];
    const float aup_i  = a_up[rbase + i];
    const float adn_i  = a_dn[rbase + i];
    const float Pi     = P[rbase + i];

    const float4 pr = *(const float4*)(prior + idx);
    float g[4], nb[4];
    #pragma unroll
    for (int t = 0; t < 4; ++t) {
        const int k = k0 + t;
        const float prk = (&pr.x)[t];
        const float aik = (unif_i > 0.5f) ? INV_S
                        : (k == i + 1 ? aup_i : (k == i - 1 ? adn_i : 0.f));
        float aki;
        if (unif[rbase + k] > 0.5f) aki = INV_S;
        else if (i == k + 1)        aki = a_up[rbase + k];
        else if (i == k - 1)        aki = a_dn[rbase + k];
        else                        aki = 0.f;
        const float nbv = prk + (1.0f - prk) * sqrtf(aik * aki + EPSV);
        nb[t] = nbv;
        const float Pk = P[rbase + k];
        if      (k > i) g[t] = __expf(Pk - Pi) + EPSV;
        else if (k < i) g[t] = __expf(Pi - Pk) + EPSV;
        else            g[t] = nbv;               // diagonal: neibor_attn[i,i]
    }
    *(float4*)(out + idx)                      = make_float4(g[0],  g[1],  g[2],  g[3]);
    *(float4*)(out + (size_t)BATCH * SS + idx) = make_float4(nb[0], nb[1], nb[2], nb[3]);
}

// =====================================================================
// Host-side launcher
// =====================================================================
extern "C" void kernel_launch(void* const* d_in, const int* in_sizes, int n_in,
                              void* d_out, int out_size, void* d_ws, size_t ws_size,
                              hipStream_t stream)
{
    const float* ctx   = (const float*)d_in[0];  // [B,S,D]
    const int*   eos   = (const int*)  d_in[1];  // [B,S,S]
    const float* prior = (const float*)d_in[2];  // [B,S,S]
    const float* Wk    = (const float*)d_in[3];  // [D,D]
    const float* bk    = (const float*)d_in[4];  // [D]
    const float* Wq    = (const float*)d_in[5];  // [D,D]
    const float* bq    = (const float*)d_in[6];  // [D]
    float* out = (float*)d_out;

    // ---- workspace layout ----
    char* w = (char*)d_ws;
    unsigned short* ctxbf = (unsigned short*)w;  w += (size_t)BS * D_MODEL * 2;      // 8 MB
    unsigned short* wqbf  = (unsigned short*)w;  w += (size_t)D_MODEL * D_MODEL * 2; // 0.5 MB
    unsigned short* wkbf  = (unsigned short*)w;  w += (size_t)D_MODEL * D_MODEL * 2; // 0.5 MB
    float* Q    = (float*)w;  w += (size_t)BS * D_MODEL * 4;                         // 16 MB
    float* K    = (float*)w;  w += (size_t)BS * D_MODEL * 4;                         // 16 MB
    float* aup  = (float*)w;  w += BS * 4;
    float* adn  = (float*)w;  w += BS * 4;
    float* unif = (float*)w;  w += BS * 4;
    float* P    = (float*)w;  w += BS * 4;

    // 0) one-shot bf16 conversions (8 floats/thread)
    cvt_f32_bf16<<<(BS * D_MODEL) / (256 * 8), 256, 0, stream>>>(ctx, (uint4*)ctxbf);
    cvt_f32_bf16<<<(D_MODEL * D_MODEL) / (256 * 8), 256, 0, stream>>>(Wq, (uint4*)wqbf);
    cvt_f32_bf16<<<(D_MODEL * D_MODEL) / (256 * 8), 256, 0, stream>>>(Wk, (uint4*)wkbf);

    // 1) projections: one block per 16-row M-tile, double-buffered async LDS
    proj_qk_wmma<<<BS / 16, 256, 0, stream>>>(ctxbf, wqbf, wkbf, bq, bk, Q, K);

    // 2) per-row neighbor softmax stats
    row_stats<<<BS / 8, 256, 0, stream>>>(Q, K, eos, aup, adn, unif);

    // 3) per-batch prefix scan
    scan_prefix<<<BATCH, SEQ, 0, stream>>>(aup, adn, unif, prior, P);

    // 4) stream both outputs
    const size_t nvec4 = (size_t)BATCH * SEQ * SEQ / 4;
    write_outputs<<<(unsigned)(nvec4 / 256), 256, 0, stream>>>(
        prior, aup, adn, unif, P, out);
}
